// SwinLocallyGroupedAttn_24644522344919
// MI455X (gfx1250) — matrix operand order repaired and checked by reading, not compiled, not run on hardware
//
#include <hip/hip_runtime.h>

// ---------------------------------------------------------------------------
// Swin shifted-window attention, fully fused per 7x7 window on gfx1250.
// One block (256 threads = 8 wave32) = one window; wave w = head w.
// All GEMMs use v_wmma_f32_16x16x32_f16 (f16 in, f32 accum).
// ---------------------------------------------------------------------------

typedef _Float16 half_t;
typedef __attribute__((ext_vector_type(16))) _Float16 v16h;
typedef __attribute__((ext_vector_type(8)))  _Float16 v8h;
typedef __attribute__((ext_vector_type(8)))  float    v8f;

#define WS_     7
#define SHIFT_  3
#define NH_     8
#define HD_     32
#define C_      256
#define H_      128
#define W_      128
#define HP_     133
#define NSIDE_  19
#define NW_     361
#define BATCH_  8
#define WSWS_   49

__device__ __forceinline__ v16h cat16(v8h lo, v8h hi) {
  return __builtin_shufflevector(lo, hi, 0,1,2,3,4,5,6,7,8,9,10,11,12,13,14,15);
}

// A operand, 16x32 f16 (ISA 7.12.2): lanes 0-15 -> M=lane, K = {k0..k0+7, k0+16..k0+23};
// lanes 16-31 -> M=lane-16, K = +8. Two aligned 16B chunks per lane.
__device__ __forceinline__ v16h load_a(const half_t* base, int stride, int m0, int k0, int lane) {
  const half_t* p = base + (m0 + (lane & 15)) * stride + k0 + ((lane >> 4) << 3);
  v8h lo = *(const v8h*)p;
  v8h hi = *(const v8h*)(p + 16);
  return cat16(lo, hi);
}

// B operand, 32x16 f16: lanes 0-15 -> N=lane, K = {k0..k0+7, k0+16..k0+23}; lanes 16-31 -> N=lane-16, K = +8.
__device__ __forceinline__ v16h load_b(const half_t* base, int stride, int n0, int k0, int lane) {
  const half_t* p = base + (n0 + (lane & 15)) * stride + k0 + ((lane >> 4) << 3);
  v8h lo = *(const v8h*)p;
  v8h hi = *(const v8h*)(p + 16);
  return cat16(lo, hi);
}

__device__ __forceinline__ v8f wmma_f16(v16h a, v16h b, v8f c) {
  return __builtin_amdgcn_wmma_f32_16x16x32_f16(false, a, false, b, (short)0, c, false, false);
}

// ---------------------------------------------------------------------------
// Prep 1: weights f32 -> f16 (L2-resident, reused by all 2888 blocks).
// ---------------------------------------------------------------------------
__global__ void cvt_weights_kernel(const float* __restrict__ qkv_w,
                                   const float* __restrict__ proj_w,
                                   half_t* __restrict__ wq16,
                                   half_t* __restrict__ wp16) {
  int i = blockIdx.x * blockDim.x + threadIdx.x;
  if (i < 3 * C_ * C_) wq16[i] = (half_t)qkv_w[i];
  if (i < C_ * C_)     wp16[i] = (half_t)proj_w[i];
}

// ---------------------------------------------------------------------------
// Prep 2: expand mask (361,49,49) -> padded (361,64,64):
//   col>=49 -> -3e38 (padded keys), row>=49 -> 0, else mask value.
// Lets the attention kernel apply scale+mask with one unconditional fma.
// ---------------------------------------------------------------------------
__global__ void build_mask_kernel(const float* __restrict__ mask,
                                  float* __restrict__ mpad) {
  int i = blockIdx.x * blockDim.x + threadIdx.x;
  if (i >= NW_ * 64 * 64) return;
  int wi  = i >> 12;
  int row = (i >> 6) & 63;
  int col = i & 63;
  float v;
  if (col >= WSWS_)      v = -3.0e38f;
  else if (row >= WSWS_) v = 0.f;
  else                   v = mask[wi * (WSWS_ * WSWS_) + row * WSWS_ + col];
  mpad[i] = v;
}

// ---------------------------------------------------------------------------
// Fused window kernel.
// LDS (132 KB, region-reused):
//   [0      , 16384) halfs : xA [64][256]    -> later P [8 heads][64][64] (lower)
//   [16384  , 32768) halfs : Q  [64][256]    -> later P upper half
//   [32768  , 49152) halfs : K  [64][256]    -> later O [64][256]
//   [49152  , 67584) halfs : Vt [256][72]    (d-major, key contiguous)
// plus a 64-entry row->output-offset scatter table.
// ---------------------------------------------------------------------------
__global__ __launch_bounds__(256, 1)
void swin_win_kernel(const float* __restrict__ x,
                     const half_t* __restrict__ wq16,
                     const half_t* __restrict__ wp16,
                     const float* __restrict__ qkv_b,
                     const float* __restrict__ proj_b,
                     const float* __restrict__ mpad,
                     float* __restrict__ out)
{
  __shared__ __align__(16) half_t smem[16384 * 3 + 256 * 72];
  __shared__ int rowOff[64];      // output linear base per window row, -1 = drop
  half_t* xA = smem;              // [64][256]
  half_t* Qs = smem + 16384;      // [64][256]
  half_t* Ks = smem + 32768;      // [64][256]
  half_t* Vt = smem + 49152;      // [256][72]
  half_t* Pb = smem;              // [8][64][64], reuses xA+Qs
  half_t* Os = smem + 32768;      // [64][256],  reuses Ks

  const int tid  = threadIdx.x;
  const int lane = tid & 31;
  const int wave = __builtin_amdgcn_readfirstlane(tid >> 5);   // scalar head id
  const int l15  = lane & 15;
  const int hi8  = (lane >> 4) << 3;

  const int widx = blockIdx.x;
  const int bb = widx / NW_;
  const int wi = widx - bb * NW_;
  const int wh = wi / NSIDE_;
  const int ww = wi - wh * NSIDE_;

  // ---- Phase 0: gather rolled/padded window rows -> LDS f16 ----------------
  {
    int r  = tid >> 2;              // 0..63
    int cq = (tid & 3) << 6;        // 0,64,128,192
    half_t* dst = xA + r * C_ + cq;
    const float* src = nullptr;
    int obase = -1;
    if (r < WSWS_) {
      int wy = r / WS_, wx = r - wy * WS_;
      int hp = wh * WS_ + wy + SHIFT_; if (hp >= HP_) hp -= HP_;
      int wp = ww * WS_ + wx + SHIFT_; if (wp >= HP_) wp -= HP_;
      if (hp < H_ && wp < W_) {
        obase = (bb * (H_ * W_) + hp * W_ + wp) * C_;
        src = x + (size_t)obase + cq;
      }
    }
    if ((tid & 3) == 0) rowOff[r] = obase;
    if (src) {
      #pragma unroll
      for (int i = 0; i < 8; ++i) {
        float4 a = ((const float4*)src)[2 * i];
        float4 b = ((const float4*)src)[2 * i + 1];
        v8h h = { (half_t)a.x, (half_t)a.y, (half_t)a.z, (half_t)a.w,
                  (half_t)b.x, (half_t)b.y, (half_t)b.z, (half_t)b.w };
        *(v8h*)(dst + i * 8) = h;
      }
    } else {
      v8h z = {};
      #pragma unroll
      for (int i = 0; i < 8; ++i) *(v8h*)(dst + i * 8) = z;
    }
  }
  __syncthreads();

  // ---- Phase 1: QKV GEMM [64,256]x[256,768]^T + bias -----------------------
  // nt is wave-uniform -> scalar routing branches (no exec juggling).
  for (int j = 0; j < 6; ++j) {
    int nt = __builtin_amdgcn_readfirstlane(wave * 6 + j);   // 0..47
    int n0 = nt << 4;
    float bias = qkv_b[n0 + l15];
    for (int mt = 0; mt < 4; ++mt) {
      v8f c = {};
      #pragma unroll
      for (int kt = 0; kt < 8; ++kt) {
        v16h a = load_a(xA, C_, mt << 4, kt << 5, lane);
        v16h b = load_b(wq16, C_, n0, kt << 5, lane);
        c = wmma_f16(a, b, c);
      }
      v8h hv;
      #pragma unroll
      for (int rr = 0; rr < 8; ++rr) hv[rr] = (half_t)(c[rr] + bias);
      int rbase = (mt << 4) + hi8;
      if (nt < 16) {                 // Q, row-major strided b16 stores
        half_t* dst = Qs + rbase * C_ + n0 + l15;
        #pragma unroll
        for (int rr = 0; rr < 8; ++rr) dst[rr * C_] = hv[rr];
      } else if (nt < 32) {          // K
        half_t* dst = Ks + rbase * C_ + (n0 - 256) + l15;
        #pragma unroll
        for (int rr = 0; rr < 8; ++rr) dst[rr * C_] = hv[rr];
      } else {                       // V transposed: 8 rows contiguous -> b128
        *(v8h*)(Vt + (n0 - 512 + l15) * 72 + rbase) = hv;
      }
    }
  }
  __syncthreads();

  // ---- Phase 2: S = Q_h K_h^T * scale + mask (16 tiles in VGPRs) -----------
  const int hoff = wave * HD_;
  v8f acc[4][4];
  #pragma unroll
  for (int mt = 0; mt < 4; ++mt) {
    v16h a = load_a(Qs + hoff, C_, mt << 4, 0, lane);
    #pragma unroll
    for (int nt = 0; nt < 4; ++nt) {
      v16h b = load_b(Ks + hoff, C_, nt << 4, 0, lane);
      v8f c = {};
      acc[mt][nt] = wmma_f16(a, b, c);
    }
  }
  const float scale = 0.17677669529663687f;       // 32^-0.5
  const float* mp = mpad + ((size_t)wi << 12);
  #pragma unroll
  for (int mt = 0; mt < 4; ++mt)
    #pragma unroll
    for (int nt = 0; nt < 4; ++nt) {
      const float* mq = mp + (((mt << 4) + hi8) << 6) + (nt << 4) + l15;
      #pragma unroll
      for (int rr = 0; rr < 8; ++rr)
        acc[mt][nt][rr] = fmaf(acc[mt][nt][rr], scale, mq[rr << 6]);
    }
  __syncthreads();   // all waves done reading Q/K -> regions reusable

  // ---- Phase 3: softmax over keys (rows live in 16-lane halves) ------------
  half_t* Ph = Pb + wave * (64 * 64);
  #pragma unroll
  for (int mt = 0; mt < 4; ++mt) {
    #pragma unroll
    for (int rr = 0; rr < 8; ++rr) {
      float m = fmaxf(fmaxf(acc[mt][0][rr], acc[mt][1][rr]),
                      fmaxf(acc[mt][2][rr], acc[mt][3][rr]));
      #pragma unroll
      for (int off = 8; off >= 1; off >>= 1) m = fmaxf(m, __shfl_xor(m, off, 32));
      float s = 0.f;
      #pragma unroll
      for (int nt = 0; nt < 4; ++nt) {
        float e = __expf(acc[mt][nt][rr] - m);
        acc[mt][nt][rr] = e;
        s += e;
      }
      #pragma unroll
      for (int off = 8; off >= 1; off >>= 1) s += __shfl_xor(s, off, 32);
      float inv = 1.f / s;
      int row = (mt << 4) + rr + hi8;
      #pragma unroll
      for (int nt = 0; nt < 4; ++nt)
        Ph[row * 64 + (nt << 4) + l15] = (half_t)(acc[mt][nt][rr] * inv);
    }
  }

  // ---- Phase 4: O_h = P_h @ V_h (K=64 -> 2 WMMA steps) ---------------------
  #pragma unroll
  for (int mt = 0; mt < 4; ++mt)
    #pragma unroll
    for (int nt = 0; nt < 2; ++nt) {
      v8f c = {};
      #pragma unroll
      for (int kt = 0; kt < 2; ++kt) {
        v16h a = load_a(Ph, 64, mt << 4, kt << 5, lane);
        v16h b = load_b(Vt + hoff * 72, 72, nt << 4, kt << 5, lane);
        c = wmma_f16(a, b, c);
      }
      half_t* dst = Os + ((mt << 4) + hi8) * C_ + hoff + (nt << 4) + l15;
      #pragma unroll
      for (int rr = 0; rr < 8; ++rr) dst[rr * C_] = (half_t)c[rr];
    }
  __syncthreads();

  // ---- Phase 5: proj GEMM + bias, table-driven reverse-roll scatter --------
  for (int j = 0; j < 2; ++j) {
    int nt = __builtin_amdgcn_readfirstlane(wave * 2 + j);   // 0..15
    int n0 = nt << 4;
    int col = n0 + l15;
    float bias = proj_b[col];
    for (int mt = 0; mt < 4; ++mt) {
      v8f c = {};
      #pragma unroll
      for (int kt = 0; kt < 8; ++kt) {
        v16h a = load_a(Os, C_, mt << 4, kt << 5, lane);
        v16h b = load_b(wp16, C_, n0, kt << 5, lane);
        c = wmma_f16(a, b, c);
      }
      #pragma unroll
      for (int rr = 0; rr < 8; ++rr) {
        int ro = rowOff[(mt << 4) + rr + hi8];
        if (ro >= 0) out[(size_t)ro + col] = c[rr] + bias;
      }
    }
  }
}

// ---------------------------------------------------------------------------
extern "C" void kernel_launch(void* const* d_in, const int* in_sizes, int n_in,
                              void* d_out, int out_size, void* d_ws, size_t ws_size,
                              hipStream_t stream) {
  (void)in_sizes; (void)n_in; (void)out_size; (void)ws_size;
  const float* x      = (const float*)d_in[0];
  const float* qkv_w  = (const float*)d_in[1];
  const float* qkv_b  = (const float*)d_in[2];
  const float* proj_w = (const float*)d_in[3];
  const float* proj_b = (const float*)d_in[4];
  const float* mask   = (const float*)d_in[5];
  float* out = (float*)d_out;

  half_t* wq16 = (half_t*)d_ws;                         // 768*256 f16 = 384 KB
  half_t* wp16 = (half_t*)d_ws + 3 * C_ * C_;           // 256*256 f16 = 128 KB
  float*  mpad = (float*)((char*)d_ws + 3 * C_ * C_ * 2 + C_ * C_ * 2);  // 361*64*64 f32

  cvt_weights_kernel<<<(3 * C_ * C_ + 255) / 256, 256, 0, stream>>>(qkv_w, proj_w, wq16, wp16);
  build_mask_kernel<<<(NW_ * 64 * 64 + 255) / 256, 256, 0, stream>>>(mask, mpad);
  swin_win_kernel<<<BATCH_ * NW_, 256, 0, stream>>>(x, wq16, wp16, qkv_b, proj_b, mpad, out);
}
